// MultiQueryAttention_43739946943118
// MI455X (gfx1250) — matrix operand order repaired and checked by reading, not compiled, not run on hardware
//
#include <hip/hip_runtime.h>
#include <hip/hip_bf16.h>

// ---------------------------------------------------------------------------
// MQA forward for MI455X (gfx1250): double-buffered TDM staging + WMMA bf16.
//   pass 0: f32 -> bf16 copies of x, Wqkv, out_w
//   pass 1: qkv_bf = x_bf @ Wqkv_bf^T + b   (GEMM, writes bf16 + V^T buffer)
//   pass 2: flash causal MQA (S^T trick), writes bf16 ctx
//   pass 3: out = ctx_bf @ out_w_bf^T + b   (GEMM, writes f32)
// Tile staging uses tensor_load_to_lds (TDM) with D# padding producing the
// conflict-free LDS pitches; stage it+1 is prefetched into the alternate
// buffer while WMMAs consume stage it (s_wait_tensorcnt 2 / 0).
// ---------------------------------------------------------------------------

typedef __attribute__((ext_vector_type(16))) __bf16 v16bf;
typedef __attribute__((ext_vector_type(8)))  __bf16 v8bf;
typedef __attribute__((ext_vector_type(4)))  __bf16 v4bf;
typedef __attribute__((ext_vector_type(8)))  float  v8f;
typedef __attribute__((ext_vector_type(4)))  unsigned int v4u;
typedef __attribute__((ext_vector_type(8)))  int    v8i;
typedef __attribute__((ext_vector_type(4)))  int    v4i;

__device__ __forceinline__ __bf16 f2bf(float f) {
  unsigned u = __builtin_bit_cast(unsigned, f);
  unsigned r = u + 0x7FFFu + ((u >> 16) & 1u);   // round-to-nearest-even
  unsigned short h = (unsigned short)(r >> 16);
  return __builtin_bit_cast(__bf16, h);
}

__device__ __forceinline__ v16bf cat16(v8bf lo, v8bf hi) {
  return __builtin_shufflevector(lo, hi, 0,1,2,3,4,5,6,7,8,9,10,11,12,13,14,15);
}

// A-fragment (16x32 bf16, MxK): lane L = row (L&15); K halves (L>>4)*8+[0..7]
// and +16 -> two b128 LDS loads.
__device__ __forceinline__ v16bf load_afrag(const __bf16* rowptr, int lq) {
  v8bf lo = *(const v8bf*)(rowptr + lq * 8);
  v8bf hi = *(const v8bf*)(rowptr + lq * 8 + 16);
  return cat16(lo, hi);
}

// B-fragment (32x16 bf16, KxN): lane L = col (L&15); K halves (L>>4)*8+[0..7]
// and +16 -> two b128 LDS loads.
__device__ __forceinline__ v16bf load_bfrag(const __bf16* rowptr, int lq) {
  v8bf lo = *(const v8bf*)(rowptr + lq * 8);
  v8bf hi = *(const v8bf*)(rowptr + lq * 8 + 16);
  return cat16(lo, hi);
}

// ---------------------------------------------------------------------------
// TDM 2D bf16 tile DMA global -> LDS with per-row LDS padding (D# ch.8).
// pad codes: interval (1<<code)*8 bytes, amount (code+1) DWORDs.
// tensor dims == tile dims -> no OOB possible.  6-arg clang-23 builtin.
// ---------------------------------------------------------------------------
__device__ __forceinline__ void tdm_load_2d(unsigned lds_off, const void* gptr,
                                            unsigned td0, unsigned td1,
                                            unsigned stride, unsigned pad_iv,
                                            unsigned pad_amt) {
  unsigned long long ga = (unsigned long long)gptr;
  v4u g0;
  g0[0] = 1u;                                        // count=1, user mode
  g0[1] = lds_off;                                   // LDS byte address
  g0[2] = (unsigned)ga;                              // global addr [31:0]
  g0[3] = (unsigned)((ga >> 32) & 0x01FFFFFFu) | (2u << 30);  // [56:32]|type=2
  v8i g1;
  g1[0] = (int)((1u << 16) |                         // data_size=1 -> 2 bytes
                (1u << 20) |                         // pad_enable
                (pad_iv << 22) | (pad_amt << 25));
  g1[1] = (int)((td0 & 0xFFFFu) << 16);              // tensor_dim0[15:0]
  g1[2] = (int)(((td0 >> 16) & 0xFFFFu) | ((td1 & 0xFFFFu) << 16));
  g1[3] = (int)(((td1 >> 16) & 0xFFFFu) | ((td0 & 0xFFFFu) << 16)); // tile_d0
  g1[4] = (int)(td1 & 0xFFFFu);                      // tile_dim1, tile_dim2=0
  g1[5] = (int)stride;                               // tensor_dim0_stride lo
  g1[6] = 0;
  g1[7] = 0;
  v4i z4 = {0, 0, 0, 0};
  v8i z8 = {0, 0, 0, 0, 0, 0, 0, 0};
  __builtin_amdgcn_tensor_load_to_lds(g0, g1, z4, z4, z8, 0);
}

__device__ __forceinline__ unsigned lds_off_of(const void* p) {
  return (unsigned)(unsigned long long)p;            // low 32 bits = LDS addr
}

// ---------------------------------------------------------------------------
__global__ __launch_bounds__(256) void f32_to_bf16(const float* __restrict__ s,
                                                   __bf16* __restrict__ d,
                                                   int n4) {
  int i = blockIdx.x * 256 + threadIdx.x;
  if (i < n4) {
    float4 v = *(const float4*)&s[i * 4];
    v4bf o;
    o[0] = f2bf(v.x); o[1] = f2bf(v.y); o[2] = f2bf(v.z); o[3] = f2bf(v.w);
    *(v4bf*)&d[i * 4] = o;
  }
}

// ---------------------------------------------------------------------------
// GEMM: 64x128 block tile, BK=32, 4 waves each 32x64 (8 WMMA / iter / wave).
// A[M,K]@B[N,K]^T + bias.  Double-buffered TDM prefetch pipeline.
// mode 0: write f32 C.  mode 1 (qkv): bf16 C for col<2176, V^T for >=.
// ---------------------------------------------------------------------------
#define GLDW 40

__global__ __launch_bounds__(128) void gemm_bf16_bt(
    const __bf16* __restrict__ A, const __bf16* __restrict__ B,
    const float* __restrict__ bias, float* __restrict__ Cf,
    __bf16* __restrict__ Cbf, __bf16* __restrict__ vT,
    int M, int N, int K, int mode)
{
  __shared__ __align__(16) __bf16 As[2][64 * GLDW];
  __shared__ __align__(16) __bf16 Bs[2][128 * GLDW];

  const int tid = threadIdx.x;
  const int lane = tid & 31, wave = tid >> 5;
  const int lh = lane & 15, lq = lane >> 4;
  const int wm = wave >> 1, wn = wave & 1;
  const unsigned bm = blockIdx.y * 64, bn = blockIdx.x * 128;
  const unsigned uK = (unsigned)K;
  const int niter = K / 32;

  const __bf16* Abase = A + bm * uK;
  const __bf16* Bbase = B + bn * uK;

  if (wave == 0) {                                   // stage 0
    tdm_load_2d(lds_off_of(As[0]), Abase, 32, 64, uK, 3, 3);
    tdm_load_2d(lds_off_of(Bs[0]), Bbase, 32, 128, uK, 3, 3);
  }

  v8f acc[2][4] = {};

  for (int it = 0; it < niter; ++it) {
    const int cur = it & 1;
    if (wave == 0) {
      if (it + 1 < niter) {                          // prefetch stage it+1
        unsigned k1 = (unsigned)(it + 1) * 32u;
        tdm_load_2d(lds_off_of(As[1 - cur]), Abase + k1, 32, 64, uK, 3, 3);
        tdm_load_2d(lds_off_of(Bs[1 - cur]), Bbase + k1, 32, 128, uK, 3, 3);
        __builtin_amdgcn_s_wait_tensorcnt(2);        // stage it complete
      } else {
        __builtin_amdgcn_s_wait_tensorcnt(0);
      }
    }
    __syncthreads();

    v16bf af[2], bfg[4];
    #pragma unroll
    for (int mt = 0; mt < 2; ++mt)
      af[mt] = load_afrag(&As[cur][(wm * 32 + mt * 16 + lh) * GLDW], lq);
    #pragma unroll
    for (int nt = 0; nt < 4; ++nt)
      bfg[nt] = load_bfrag(&Bs[cur][(wn * 64 + nt * 16 + lh) * GLDW], lq);

    #pragma unroll
    for (int mt = 0; mt < 2; ++mt)
      #pragma unroll
      for (int nt = 0; nt < 4; ++nt)
        acc[mt][nt] = __builtin_amdgcn_wmma_f32_16x16x32_bf16(
            false, af[mt], false, bfg[nt], (short)0, acc[mt][nt], false, false);

    __syncthreads();
  }

  #pragma unroll
  for (int nt = 0; nt < 4; ++nt) {
    unsigned col = bn + wn * 64 + nt * 16 + lh;
    float bv = bias[col];
    #pragma unroll
    for (int mt = 0; mt < 2; ++mt) {
      unsigned rowbase = bm + wm * 32 + mt * 16 + 8 * lq;
      if (mode == 0) {
        #pragma unroll
        for (int r = 0; r < 8; ++r)
          Cf[(rowbase + r) * (unsigned)N + col] = acc[mt][nt][r] + bv;
      } else if (col < 2176u) {                      // Q and K columns
        #pragma unroll
        for (int r = 0; r < 8; ++r)
          Cbf[(rowbase + r) * (unsigned)N + col] = f2bf(acc[mt][nt][r] + bv);
      } else {                                       // V columns -> V^T [128,M]
        v8bf pk;
        #pragma unroll
        for (int r = 0; r < 8; ++r) pk[r] = f2bf(acc[mt][nt][r] + bv);
        *(v8bf*)&vT[(col - 2176u) * (unsigned)M + rowbase] = pk;
      }
    }
  }
}

// ---------------------------------------------------------------------------
// Flash causal MQA on bf16 qkv.  Grid (S/64, HEADS), 128 thr = 4 waves.
// S^T = K.Q^T so score C-layout == P A-fragment layout (no LDS transpose);
// V comes pre-transposed (vT) so PV B-fragments are contiguous b128 reads.
// K/V tiles double-buffered, next block prefetched while computing current.
// ---------------------------------------------------------------------------
#define QLDW 136
#define KLDW 136
#define VLDW 40

__global__ __launch_bounds__(128) void mqa_attn(
    const __bf16* __restrict__ qkvb, const __bf16* __restrict__ vT,
    __bf16* __restrict__ ctxb)
{
  __shared__ __align__(16) __bf16 Qs[64 * QLDW];
  __shared__ __align__(16) __bf16 Ks[2][32 * KLDW];
  __shared__ __align__(16) __bf16 Vs[2][128 * VLDW]; // [dcol][key]

  const int tid = threadIdx.x;
  const int lane = tid & 31, wave = tid >> 5;
  const int lh = lane & 15, lq = lane >> 4;
  const unsigned head = blockIdx.y;
  const unsigned qb = blockIdx.x * 64;
  const unsigned LD = 2304;

  const int nkb = ((int)qb + 63) / 32 + 1;

  if (wave == 0) {
    tdm_load_2d(lds_off_of(Qs), qkvb + qb * LD + head * 128u, 128, 64, LD, 5, 3);
    tdm_load_2d(lds_off_of(Ks[0]), qkvb + 2048u, 128, 32, LD, 5, 3);
    tdm_load_2d(lds_off_of(Vs[0]), vT, 32, 128, 2048, 3, 3);
    __builtin_amdgcn_s_wait_tensorcnt(0);            // Q + stage 0 ready
  }
  __syncthreads();

  v16bf qf[4];
  #pragma unroll
  for (int c = 0; c < 4; ++c)
    qf[c] = load_bfrag(&Qs[(wave * 16 + lh) * QLDW + c * 32], lq);

  v8f o[8] = {};
  float m_i = -1e30f, l_i = 0.f;
  const float sc = 0.08838834764831845f;             // 1/sqrt(128)
  const int qmaxw = (int)qb + wave * 16 + 15;        // wave-uniform
  const int my_q  = (int)qb + wave * 16 + lh;

  for (int kb = 0; kb < nkb; ++kb) {
    const int kbase = kb * 32;
    const int cur = kb & 1;

    if (wave == 0) {
      if (kb + 1 < nkb) {                            // prefetch stage kb+1
        unsigned nb = (unsigned)(kbase + 32);
        tdm_load_2d(lds_off_of(Ks[1 - cur]), qkvb + nb * LD + 2048u,
                    128, 32, LD, 5, 3);
        tdm_load_2d(lds_off_of(Vs[1 - cur]), vT + nb, 32, 128, 2048, 3, 3);
        __builtin_amdgcn_s_wait_tensorcnt(2);        // stage kb complete
      } else {
        __builtin_amdgcn_s_wait_tensorcnt(0);
      }
    }
    __syncthreads();

    if (kbase <= qmaxw) {                            // wave-uniform
      v8f st[2] = {};
      #pragma unroll
      for (int s = 0; s < 2; ++s)
        #pragma unroll
        for (int c = 0; c < 4; ++c) {
          v16bf kf = load_afrag(&Ks[cur][(s * 16 + lh) * KLDW + c * 32], lq);
          st[s] = __builtin_amdgcn_wmma_f32_16x16x32_bf16(
              false, kf, false, qf[c], (short)0, st[s], false, false);
        }

      float rmax = -1e30f;
      #pragma unroll
      for (int s = 0; s < 2; ++s)
        #pragma unroll
        for (int r = 0; r < 8; ++r) {
          int key = kbase + s * 16 + r + 8 * lq;
          float v = st[s][r] * sc;
          v = (key <= my_q) ? v : -1e30f;
          st[s][r] = v;
          rmax = fmaxf(rmax, v);
        }
      rmax = fmaxf(rmax, __shfl_xor(rmax, 16, 32));
      float mnew = fmaxf(m_i, rmax);

      float rsum = 0.f;
      #pragma unroll
      for (int s = 0; s < 2; ++s)
        #pragma unroll
        for (int r = 0; r < 8; ++r) {
          float p = __expf(st[s][r] - mnew);
          st[s][r] = p;
          rsum += p;
        }
      rsum += __shfl_xor(rsum, 16, 32);

      float corr = __expf(m_i - mnew);
      l_i = l_i * corr + rsum;
      m_i = mnew;

      #pragma unroll
      for (int r = 0; r < 8; ++r) {
        float cr = __shfl(corr, r + 8 * lq, 32);
        #pragma unroll
        for (int t = 0; t < 8; ++t) o[t][r] *= cr;
      }

      v16bf pf;                                      // P A-frag from S^T tiles
      #pragma unroll
      for (int j = 0; j < 8; ++j) {
        pf[j]     = f2bf(st[0][j]);
        pf[j + 8] = f2bf(st[1][j]);
      }

      #pragma unroll
      for (int t = 0; t < 8; ++t) {
        v16bf vf = load_bfrag(&Vs[cur][(t * 16 + lh) * VLDW], lq);
        o[t] = __builtin_amdgcn_wmma_f32_16x16x32_bf16(
            false, pf, false, vf, (short)0, o[t], false, false);
      }
    }
    __syncthreads();
  }

  float linv = 1.f / l_i;
  #pragma unroll
  for (int r = 0; r < 8; ++r) {
    float lr = __shfl(linv, r + 8 * lq, 32);
    unsigned row = qb + wave * 16 + r + 8 * lq;
    #pragma unroll
    for (int t = 0; t < 8; ++t)
      ctxb[row * 2048u + head * 128u + t * 16u + lh] = f2bf(o[t][r] * lr);
  }
}

// ---------------------------------------------------------------------------
extern "C" void kernel_launch(void* const* d_in, const int* in_sizes, int n_in,
                              void* d_out, int out_size, void* d_ws, size_t ws_size,
                              hipStream_t stream) {
  (void)in_sizes; (void)n_in; (void)out_size; (void)ws_size;
  const float* x      = (const float*)d_in[0];   // [2048, 2048]
  const float* Wqkv_w = (const float*)d_in[1];   // [2304, 2048]
  const float* Wqkv_b = (const float*)d_in[2];   // [2304]
  const float* out_w  = (const float*)d_in[3];   // [2048, 2048]
  const float* out_b  = (const float*)d_in[4];   // [2048]
  float* out = (float*)d_out;

  const unsigned NX = 2048u * 2048u, NW1 = 2304u * 2048u;
  __bf16* x_bf   = (__bf16*)d_ws;
  __bf16* w1_bf  = x_bf   + NX;                  // Wqkv bf16
  __bf16* w2_bf  = w1_bf  + NW1;                 // out_w bf16
  __bf16* qkv_bf = w2_bf  + NX;                  // [2048, 2304]
  __bf16* vT     = qkv_bf + 2048u * 2304u;       // [128, 2048]
  __bf16* ctx_bf = vT     + 128u * 2048u;        // [2048, 2048]

  f32_to_bf16<<<NX  / 4 / 256, 256, 0, stream>>>(x,      x_bf,  NX / 4);
  f32_to_bf16<<<NW1 / 4 / 256, 256, 0, stream>>>(Wqkv_w, w1_bf, NW1 / 4);
  f32_to_bf16<<<NX  / 4 / 256, 256, 0, stream>>>(out_w,  w2_bf, NX / 4);

  gemm_bf16_bt<<<dim3(2304 / 128, 2048 / 64), 128, 0, stream>>>(
      x_bf, w1_bf, Wqkv_b, nullptr, qkv_bf, vT, 2048, 2304, 2048, 1);
  mqa_attn<<<dim3(2048 / 64, 16), 128, 0, stream>>>(qkv_bf, vT, ctx_bf);
  gemm_bf16_bt<<<dim3(2048 / 128, 2048 / 64), 128, 0, stream>>>(
      ctx_bf, w2_bf, out_b, out, nullptr, nullptr, 2048, 2048, 2048, 0);
}
